// TransformerCritic_40510131536193
// MI455X (gfx1250) — matrix operand order repaired and checked by reading, not compiled, not run on hardware
//
#include <hip/hip_runtime.h>
#include <math.h>

// ---------------- types ----------------
typedef __attribute__((ext_vector_type(16))) __bf16 v16bf;
typedef __attribute__((ext_vector_type(8)))  float  v8f;

#define DEV static __device__ __forceinline__

struct __align__(16) U4 { unsigned x, y, z, w; };
struct __align__(16) F4 { float x, y, z, w; };

union Frag {
  v16bf v;
  U4 q[2];
  __bf16 h[16];
};

// ---------------- problem constants ----------------
#define HDIM   64
#define DO     128
#define NA     16
#define AGENTS 8
#define TT     512
#define NROW   131072   // B*T*A
#define NTOK   16384    // B*T

// ---------------- workspace layout (bf16 elements) ----------------
#define OFF_SE  0        // 64 x 128
#define OFF_SA  8192     // 64 x 160 (K=144 zero-padded)
#define OFF_KEY 18432    // 64 x 64
#define OFF_QRY 22528
#define OFF_VAL 26624
#define OFF_L1  30720
#define OFF_L2  34816
#define OFF_QW  38912    // 64 x 128
#define OFF_CW  47104    // 64 x 128
#define OFF_IH  55296    // 192 x 64
#define OFF_HH  67584    // 192 x 64
#define OFF_NF  81920    // 131072 x 64 bf16, (B*A, T, H) layout

// ---------------- helpers ----------------
DEV float gelu_f(float x){ return 0.5f * x * (1.f + erff(x * 0.70710678118654752f)); }
DEV float red16(float x){
  x += __shfl_xor(x, 1, 32); x += __shfl_xor(x, 2, 32);
  x += __shfl_xor(x, 4, 32); x += __shfl_xor(x, 8, 32);
  return x;
}
DEV v8f vzero(){
  v8f z;
  #pragma unroll
  for(int i = 0; i < 8; i++) z[i] = 0.f;
  return z;
}
DEV v8f mma(const Frag& a, const Frag& b, v8f c){
  return __builtin_amdgcn_wmma_f32_16x16x32_bf16(false, a.v, false, b.v, (short)0, c, false, false);
}

// A-fragment (16x32 bf16) from row-major bf16 tile.
// lanes 0-15: M=lane, K in [0,8)∪[16,24); lanes 16-31: M=lane-16, K in [8,16)∪[24,32)
DEV Frag afrag_bf(const __bf16* t, int stride, int kt, int lane){
  int row = lane & 15;
  int o = kt * 32 + ((lane < 16) ? 0 : 8);
  const __bf16* p = t + (size_t)row * stride + o;
  Frag f;
  f.q[0] = *(const U4*)p;
  f.q[1] = *(const U4*)(p + 16);
  return f;
}
// A-fragment built from row-major f32 (native f32->bf16 cvt in regs)
DEV Frag afrag_f32(const float* t, int stride, int kt, int lane){
  int row = lane & 15;
  int o = kt * 32 + ((lane < 16) ? 0 : 8);
  const float* p = t + (size_t)row * stride + o;
  F4 a = *(const F4*)p;         F4 b = *(const F4*)(p + 4);
  F4 cc = *(const F4*)(p + 16); F4 d = *(const F4*)(p + 20);
  Frag f;
  f.h[0]  = (__bf16)a.x;  f.h[1]  = (__bf16)a.y;  f.h[2]  = (__bf16)a.z;  f.h[3]  = (__bf16)a.w;
  f.h[4]  = (__bf16)b.x;  f.h[5]  = (__bf16)b.y;  f.h[6]  = (__bf16)b.z;  f.h[7]  = (__bf16)b.w;
  f.h[8]  = (__bf16)cc.x; f.h[9]  = (__bf16)cc.y; f.h[10] = (__bf16)cc.z; f.h[11] = (__bf16)cc.w;
  f.h[12] = (__bf16)d.x;  f.h[13] = (__bf16)d.y;  f.h[14] = (__bf16)d.z;  f.h[15] = (__bf16)d.w;
  return f;
}
// actions k-tile: 16 real features, zero-padded to K=32
DEV Frag afrag_act(const float* t, int lane){
  int row = lane & 15;
  int o = (lane < 16) ? 0 : 8;
  const float* p = t + (size_t)row * NA + o;
  F4 a = *(const F4*)p; F4 b = *(const F4*)(p + 4);
  Frag f;
  f.h[0] = (__bf16)a.x; f.h[1] = (__bf16)a.y; f.h[2] = (__bf16)a.z; f.h[3] = (__bf16)a.w;
  f.h[4] = (__bf16)b.x; f.h[5] = (__bf16)b.y; f.h[6] = (__bf16)b.z; f.h[7] = (__bf16)b.w;
  f.q[1] = U4{0,0,0,0};
  return f;
}
// B-fragment (32x16 bf16) from [n][k] layout (pre-transposed weights, or K rows for Q@K^T):
// lanes 0-15: N=lane, K in [0,8)∪[16,24); lanes 16-31: N=lane-16, K in [8,16)∪[24,32)
DEV Frag bfrag(const __bf16* wt, int Kpad, int nt, int kt, int lane){
  int n = lane & 15;
  int o = kt * 32 + ((lane < 16) ? 0 : 8);
  const __bf16* p = wt + (size_t)(nt * 16 + n) * Kpad + o;
  Frag f;
  f.q[0] = *(const U4*)p;
  f.q[1] = *(const U4*)(p + 16);
  return f;
}
// B-fragment from V^T (64 x 16): K=0..15 real (rows of V), upper K zero-padded
DEV Frag bfrag_vt(const __bf16* vt, int nt, int lane){
  Frag f;
  {
    const __bf16* p = vt + (size_t)(nt * 16 + (lane & 15)) * 16 + ((lane < 16) ? 0 : 8);
    f.q[0] = *(const U4*)p;
    f.q[1] = U4{0,0,0,0};
  }
  return f;
}
// A-fragment from 16x16 attention-weight tile, K zero-padded to 32
DEV Frag afrag_w8(const __bf16* wtile, int lane){
  int row = lane & 15;
  int o = (lane < 16) ? 0 : 8;
  Frag f;
  f.q[0] = *(const U4*)(wtile + row * 16 + o);
  f.q[1] = U4{0,0,0,0};
  return f;
}

// ---- epilogues on a 16x64 C-tile set (c[4] v8f) ----
DEV void epi_bias_gelu(v8f* c, const float* bias, int l15){
  #pragma unroll
  for(int nt = 0; nt < 4; nt++){
    float bb = bias[nt * 16 + l15];
    #pragma unroll
    for(int r = 0; r < 8; r++) c[nt][r] = gelu_f(c[nt][r] + bb);
  }
}
DEV void epi_ln(v8f* c, const float* g, const float* be, int l15){
  float gg[4], bb[4];
  #pragma unroll
  for(int nt = 0; nt < 4; nt++){ gg[nt] = g[nt * 16 + l15]; bb[nt] = be[nt * 16 + l15]; }
  #pragma unroll
  for(int r = 0; r < 8; r++){
    float s = 0.f;
    #pragma unroll
    for(int nt = 0; nt < 4; nt++) s += c[nt][r];
    float m = red16(s) * (1.f / 64.f);
    float v = 0.f;
    #pragma unroll
    for(int nt = 0; nt < 4; nt++){ float d = c[nt][r] - m; v += d * d; }
    float rstd = rsqrtf(red16(v) * (1.f / 64.f) + 1e-5f);
    #pragma unroll
    for(int nt = 0; nt < 4; nt++) c[nt][r] = (c[nt][r] - m) * rstd * gg[nt] + bb[nt];
  }
}
DEV void store_tile(__bf16* t, const v8f* c, int lane){
  int l15 = lane & 15; int mh = (lane < 16) ? 0 : 8;
  #pragma unroll
  for(int nt = 0; nt < 4; nt++){
    #pragma unroll
    for(int r = 0; r < 8; r++)
      t[(r + mh) * 64 + nt * 16 + l15] = (__bf16)c[nt][r];
  }
}

// ---------------- prep: transpose weights to bf16 [n][k] ----------------
DEV void wtrans(__bf16* dst, const float* W, int K, int N, int Kpad, int tid, int nth){
  for(int i = tid; i < N * Kpad; i += nth){
    int n = i / Kpad, k = i - n * Kpad;
    dst[i] = (k < K) ? (__bf16)W[(size_t)k * N + n] : (__bf16)0.f;
  }
}
__global__ __launch_bounds__(256) void prep_kernel(
    const float* seW, const float* saW, const float* keyW, const float* qryW,
    const float* valW, const float* l1W, const float* l2W, const float* qW,
    const float* cW, const float* ihW, const float* hhW, __bf16* ws){
  int tid = blockIdx.x * blockDim.x + threadIdx.x;
  int nth = gridDim.x * blockDim.x;
  wtrans(ws + OFF_SE,  seW, 128,  64, 128, tid, nth);
  wtrans(ws + OFF_SA,  saW, 144,  64, 160, tid, nth);
  wtrans(ws + OFF_KEY, keyW, 64,  64,  64, tid, nth);
  wtrans(ws + OFF_QRY, qryW, 64,  64,  64, tid, nth);
  wtrans(ws + OFF_VAL, valW, 64,  64,  64, tid, nth);
  wtrans(ws + OFF_L1,  l1W,  64,  64,  64, tid, nth);
  wtrans(ws + OFF_L2,  l2W,  64,  64,  64, tid, nth);
  wtrans(ws + OFF_QW,  qW,  128,  64, 128, tid, nth);
  wtrans(ws + OFF_CW,  cW,  128,  64, 128, tid, nth);
  wtrans(ws + OFF_IH,  ihW,  64, 192,  64, tid, nth);
  wtrans(ws + OFF_HH,  hhW,  64, 192,  64, tid, nth);
}

// ---------------- fused embed + attention + MLP + concat kernel ----------------
__global__ __launch_bounds__(256) void attn_kernel(
    const float* __restrict__ states, const float* __restrict__ actions,
    const float* sa_b, const float* sa_g, const float* sa_be,
    const float* se_b, const float* se_g, const float* se_be,
    const float* av_g, const float* av_be,
    const float* l1_b, const float* ln_g, const float* ln_be,
    const float* l2_b, const float* avl_g, const float* avl_be,
    const float* q_b, const float* q_g, const float* q_be,
    const float* c_b,
    const __bf16* __restrict__ wt,
    __bf16* __restrict__ nf_ws,
    float* __restrict__ w_out)
{
  __shared__ __align__(16) __bf16 sm[8][3328];   // per-wave: bufA, bufB, V^T, W8
  const int w = threadIdx.x >> 5, lane = threadIdx.x & 31, l15 = lane & 15;
  const int mh = (lane < 16) ? 0 : 8;
  __bf16* bufA  = &sm[w][0];       // 16x64
  __bf16* bufB  = bufA + 1024;     // 16x64
  __bf16* bufVT = bufA + 2048;     // 64x16
  __bf16* bufW8 = bufA + 3072;     // 16x16
  const size_t Rbase = (size_t)blockIdx.x * 128 + (size_t)w * 16; // 16 rows = 2 tokens
  const float* sbase = states  + Rbase * DO;
  const float* abase = actions + Rbase * NA;
  v8f c[4];

  // ---- se = LN(gelu(s @ se_W + b)) -> bufA ----
  #pragma unroll
  for(int nt = 0; nt < 4; nt++) c[nt] = vzero();
  #pragma unroll
  for(int kt = 0; kt < 4; kt++){
    Frag a = afrag_f32(sbase, DO, kt, lane);
    #pragma unroll
    for(int nt = 0; nt < 4; nt++){ Frag b = bfrag(wt + OFF_SE, 128, nt, kt, lane); c[nt] = mma(a, b, c[nt]); }
  }
  epi_bias_gelu(c, se_b, l15); epi_ln(c, se_g, se_be, l15); store_tile(bufA, c, lane);

  // ---- sae = LN(gelu([s,a] @ sa_W + b)) -> bufB ----
  #pragma unroll
  for(int nt = 0; nt < 4; nt++) c[nt] = vzero();
  #pragma unroll
  for(int kt = 0; kt < 4; kt++){
    Frag a = afrag_f32(sbase, DO, kt, lane);
    #pragma unroll
    for(int nt = 0; nt < 4; nt++){ Frag b = bfrag(wt + OFF_SA, 160, nt, kt, lane); c[nt] = mma(a, b, c[nt]); }
  }
  {
    Frag a = afrag_act(abase, lane);
    #pragma unroll
    for(int nt = 0; nt < 4; nt++){ Frag b = bfrag(wt + OFF_SA, 160, nt, 4, lane); c[nt] = mma(a, b, c[nt]); }
  }
  epi_bias_gelu(c, sa_b, l15); epi_ln(c, sa_g, sa_be, l15); store_tile(bufB, c, lane);

  // ---- V = sae @ val_W, stored transposed (bufVT[h][j]) ----
  #pragma unroll
  for(int nt = 0; nt < 4; nt++) c[nt] = vzero();
  #pragma unroll
  for(int kt = 0; kt < 2; kt++){
    Frag a = afrag_bf(bufB, 64, kt, lane);
    #pragma unroll
    for(int nt = 0; nt < 4; nt++){ Frag b = bfrag(wt + OFF_VAL, 64, nt, kt, lane); c[nt] = mma(a, b, c[nt]); }
  }
  #pragma unroll
  for(int nt = 0; nt < 4; nt++){
    #pragma unroll
    for(int r = 0; r < 8; r++) bufVT[(nt * 16 + l15) * 16 + (r + mh)] = (__bf16)c[nt][r];
  }

  // ---- K = se @ key_W -> bufB (sae no longer needed) ----
  #pragma unroll
  for(int nt = 0; nt < 4; nt++) c[nt] = vzero();
  #pragma unroll
  for(int kt = 0; kt < 2; kt++){
    Frag a = afrag_bf(bufA, 64, kt, lane);
    #pragma unroll
    for(int nt = 0; nt < 4; nt++){ Frag b = bfrag(wt + OFF_KEY, 64, nt, kt, lane); c[nt] = mma(a, b, c[nt]); }
  }
  store_tile(bufB, c, lane);

  // ---- Q = se @ query_W -> bufA (se no longer needed) ----
  #pragma unroll
  for(int nt = 0; nt < 4; nt++) c[nt] = vzero();
  #pragma unroll
  for(int kt = 0; kt < 2; kt++){
    Frag a = afrag_bf(bufA, 64, kt, lane);
    #pragma unroll
    for(int nt = 0; nt < 4; nt++){ Frag b = bfrag(wt + OFF_QRY, 64, nt, kt, lane); c[nt] = mma(a, b, c[nt]); }
  }
  store_tile(bufA, c, lane);

  // ---- S = Q @ K^T (16x16, block-diagonal by token) ----
  v8f cs = vzero();
  #pragma unroll
  for(int kt = 0; kt < 2; kt++){
    Frag a = afrag_bf(bufA, 64, kt, lane);  // Q rows
    Frag b = bfrag(bufB, 64, 0, kt, lane);  // K rows used as B columns
    cs = mma(a, b, cs);
  }

  // ---- masked softmax over the 7 off-diagonal neighbors; emit weight ----
  const size_t tok0 = Rbase >> 3;
  #pragma unroll
  for(int r = 0; r < 8; r++){
    float sc = cs[r] * 0.125f;                        // / sqrt(64)
    bool vgrp = (lane < 16) ? (l15 < 8) : (l15 >= 8); // same-token column
    int j = (lane < 16) ? l15 : (l15 - 8);
    bool use = vgrp && (j != r);
    float vm = use ? sc : -3.4e38f;
    vm = fmaxf(vm, __shfl_xor(vm, 1, 32));
    vm = fmaxf(vm, __shfl_xor(vm, 2, 32));
    vm = fmaxf(vm, __shfl_xor(vm, 4, 32));
    float e = use ? __expf(sc - vm) : 0.f;
    float su = e;
    su += __shfl_xor(su, 1, 32); su += __shfl_xor(su, 2, 32); su += __shfl_xor(su, 4, 32);
    float wv = use ? (e / su) : 0.f;
    if(use){
      size_t tk = tok0 + ((lane < 16) ? 0 : 1);
      int pos = (j < r) ? j : (j - 1);
      w_out[tk * (AGENTS * 7) + (size_t)r * 7 + pos] = wv;
    }
    bufW8[(r + mh) * 16 + l15] = (__bf16)wv;
  }

  // ---- X = W8 @ V via WMMA (K zero-padded) ----
  v8f cx[4];
  #pragma unroll
  for(int nt = 0; nt < 4; nt++) cx[nt] = vzero();
  {
    Frag a = afrag_w8(bufW8, lane);
    #pragma unroll
    for(int nt = 0; nt < 4; nt++){ Frag b = bfrag_vt(bufVT, nt, lane); cx[nt] = mma(a, b, cx[nt]); }
  }
  epi_ln(cx, av_g, av_be, l15);
  v8f xres[4];
  #pragma unroll
  for(int nt = 0; nt < 4; nt++) xres[nt] = cx[nt];
  store_tile(bufB, cx, lane);     // x (post av-LN)

  // ---- lin1 ----
  #pragma unroll
  for(int nt = 0; nt < 4; nt++) c[nt] = vzero();
  #pragma unroll
  for(int kt = 0; kt < 2; kt++){
    Frag a = afrag_bf(bufB, 64, kt, lane);
    #pragma unroll
    for(int nt = 0; nt < 4; nt++){ Frag b = bfrag(wt + OFF_L1, 64, nt, kt, lane); c[nt] = mma(a, b, c[nt]); }
  }
  epi_bias_gelu(c, l1_b, l15); epi_ln(c, ln_g, ln_be, l15); store_tile(bufA, c, lane);

  // ---- lin2 + residual + LN(avl) -> bufB (x final) ----
  #pragma unroll
  for(int nt = 0; nt < 4; nt++) c[nt] = vzero();
  #pragma unroll
  for(int kt = 0; kt < 2; kt++){
    Frag a = afrag_bf(bufA, 64, kt, lane);
    #pragma unroll
    for(int nt = 0; nt < 4; nt++){ Frag b = bfrag(wt + OFF_L2, 64, nt, kt, lane); c[nt] = mma(a, b, c[nt]); }
  }
  #pragma unroll
  for(int nt = 0; nt < 4; nt++){
    float bb = l2_b[nt * 16 + l15];
    #pragma unroll
    for(int r = 0; r < 8; r++) c[nt][r] += bb + xres[nt][r];
  }
  epi_ln(c, avl_g, avl_be, l15); store_tile(bufB, c, lane);

  // ---- qe = LN(gelu(s @ q_W + b)) -> bufA ----
  #pragma unroll
  for(int nt = 0; nt < 4; nt++) c[nt] = vzero();
  #pragma unroll
  for(int kt = 0; kt < 4; kt++){
    Frag a = afrag_f32(sbase, DO, kt, lane);
    #pragma unroll
    for(int nt = 0; nt < 4; nt++){ Frag b = bfrag(wt + OFF_QW, 128, nt, kt, lane); c[nt] = mma(a, b, c[nt]); }
  }
  epi_bias_gelu(c, q_b, l15); epi_ln(c, q_g, q_be, l15); store_tile(bufA, c, lane);

  // ---- nf = gelu([qe, x] @ c_W + b) ----
  #pragma unroll
  for(int nt = 0; nt < 4; nt++) c[nt] = vzero();
  #pragma unroll
  for(int kt = 0; kt < 4; kt++){
    Frag a = (kt < 2) ? afrag_bf(bufA, 64, kt, lane) : afrag_bf(bufB, 64, kt - 2, lane);
    #pragma unroll
    for(int nt = 0; nt < 4; nt++){ Frag b = bfrag(wt + OFF_CW, 128, nt, kt, lane); c[nt] = mma(a, b, c[nt]); }
  }
  epi_bias_gelu(c, c_b, l15);

  // ---- scatter nf (bf16) into (B*A, T, H) layout for the GRU ----
  #pragma unroll
  for(int nt = 0; nt < 4; nt++){
    #pragma unroll
    for(int r = 0; r < 8; r++){
      size_t n = Rbase + (size_t)(r + mh);
      size_t token = n >> 3; int ag = (int)(n & 7);
      size_t bb = token >> 9; int tt = (int)(token & 511);
      size_t grow = bb * AGENTS + ag;
      nf_ws[(grow * TT + tt) * 64 + nt * 16 + l15] = (__bf16)c[nt][r];
    }
  }
}

// ---------------- GRU + value head: 16 WGs x 1 wave, 16 sequences each ----------------
__global__ __launch_bounds__(32) void gru_kernel(
    const __bf16* __restrict__ nf_ws, const __bf16* __restrict__ wt,
    const float* __restrict__ h0, const float* bih, const float* bhh,
    const float* f_g, const float* f_be, const float* f_W, const float* f_b,
    float* __restrict__ val_out, float* __restrict__ hT_out)
{
  __shared__ __align__(16) __bf16 sWih[192 * 64];
  __shared__ __align__(16) __bf16 sWhh[192 * 64];
  __shared__ __align__(16) __bf16 hbuf[16 * 64];
  const int lane = threadIdx.x & 31, l15 = lane & 15;
  const int mh = (lane < 16) ? 0 : 8;
  const int base = blockIdx.x * 16;

  // stage both GRU weight matrices in LDS (48 KB) via CDNA5 async global->LDS DMA
  {
    const __bf16* gih = wt + OFF_IH;
    const __bf16* ghh = wt + OFF_HH;
    for(int i = lane * 8; i < 12288; i += 32 * 8){
      unsigned ldsIh = (unsigned)(size_t)&sWih[i];
      unsigned ldsHh = (unsigned)(size_t)&sWhh[i];
      unsigned long long gaIh = (unsigned long long)(size_t)(gih + i);
      unsigned long long gaHh = (unsigned long long)(size_t)(ghh + i);
      asm volatile("global_load_async_to_lds_b128 %0, %1, off"
                   :: "v"(ldsIh), "v"(gaIh) : "memory");
      asm volatile("global_load_async_to_lds_b128 %0, %1, off"
                   :: "v"(ldsHh), "v"(gaHh) : "memory");
    }
    asm volatile("s_wait_asynccnt 0" ::: "memory");
  }

  float bih_l[12], bhh_l[12], fg_l[4], fb_l[4], fw_l[4];
  #pragma unroll
  for(int nt = 0; nt < 12; nt++){ bih_l[nt] = bih[nt * 16 + l15]; bhh_l[nt] = bhh[nt * 16 + l15]; }
  #pragma unroll
  for(int nt = 0; nt < 4; nt++){
    fg_l[nt] = f_g[nt * 16 + l15]; fb_l[nt] = f_be[nt * 16 + l15]; fw_l[nt] = f_W[nt * 16 + l15];
  }
  const float fb0 = f_b[0];

  v8f hc[4];
  #pragma unroll
  for(int nt = 0; nt < 4; nt++){
    #pragma unroll
    for(int r = 0; r < 8; r++){
      int M = r + mh;
      float hv = h0[(size_t)(base + M) * 64 + nt * 16 + l15];
      hc[nt][r] = hv;
      hbuf[M * 64 + nt * 16 + l15] = (__bf16)hv;
    }
  }

  for(int t = 0; t < TT; t++){
    // A-fragments of nf rows at time t
    Frag an[2];
    #pragma unroll
    for(int kt = 0; kt < 2; kt++){
      const __bf16* p = nf_ws + ((size_t)(base + l15) * TT + t) * 64 + kt * 32 + ((lane < 16) ? 0 : 8);
      an[kt].q[0] = *(const U4*)p;
      an[kt].q[1] = *(const U4*)(p + 16);
    }
    v8f cg[12], ch[12];
    #pragma unroll
    for(int nt = 0; nt < 12; nt++){ cg[nt] = vzero(); ch[nt] = vzero(); }
    #pragma unroll
    for(int kt = 0; kt < 2; kt++){
      Frag ah = afrag_bf(hbuf, 64, kt, lane);
      #pragma unroll
      for(int nt = 0; nt < 12; nt++){
        Frag b = bfrag(sWhh, 64, nt, kt, lane);
        ch[nt] = mma(ah, b, ch[nt]);
      }
      #pragma unroll
      for(int nt = 0; nt < 12; nt++){
        Frag b = bfrag(sWih, 64, nt, kt, lane);
        cg[nt] = mma(an[kt], b, cg[nt]);
      }
    }
    // gates
    #pragma unroll
    for(int g0 = 0; g0 < 4; g0++){
      #pragma unroll
      for(int r = 0; r < 8; r++){
        float ir  = cg[g0][r]     + bih_l[g0],     hr = ch[g0][r]     + bhh_l[g0];
        float iz  = cg[g0 + 4][r] + bih_l[g0 + 4], hz = ch[g0 + 4][r] + bhh_l[g0 + 4];
        float in_ = cg[g0 + 8][r] + bih_l[g0 + 8], hn = ch[g0 + 8][r] + bhh_l[g0 + 8];
        float rg = 1.f / (1.f + __expf(-(ir + hr)));
        float zg = 1.f / (1.f + __expf(-(iz + hz)));
        float ng = tanhf(in_ + rg * hn);
        hc[g0][r] = (1.f - zg) * ng + zg * hc[g0][r];
      }
    }
    // fused value head: Value = LN(h, f_g, f_beta) @ f_W + f_b
    #pragma unroll
    for(int r = 0; r < 8; r++){
      float s = 0.f;
      #pragma unroll
      for(int nt = 0; nt < 4; nt++) s += hc[nt][r];
      float m = red16(s) * (1.f / 64.f);
      float v = 0.f;
      #pragma unroll
      for(int nt = 0; nt < 4; nt++){ float d = hc[nt][r] - m; v += d * d; }
      float rstd = rsqrtf(red16(v) * (1.f / 64.f) + 1e-5f);
      float vp = 0.f;
      #pragma unroll
      for(int nt = 0; nt < 4; nt++){
        float nv = (hc[nt][r] - m) * rstd * fg_l[nt] + fb_l[nt];
        vp += nv * fw_l[nt];
      }
      float vv = red16(vp) + fb0;
      if(l15 == 0){
        int M = r + mh; int grow = base + M;
        int bb = grow >> 3, ag = grow & 7;
        val_out[((size_t)bb * TT + t) * AGENTS + ag] = vv;
      }
    }
    // refresh bf16 h tile for next step's A-fragments
    #pragma unroll
    for(int nt = 0; nt < 4; nt++){
      #pragma unroll
      for(int r = 0; r < 8; r++){
        int M = r + mh;
        hbuf[M * 64 + nt * 16 + l15] = (__bf16)hc[nt][r];
      }
    }
    if(t + 1 < TT)
      __builtin_prefetch(nf_ws + ((size_t)(base + l15) * TT + (t + 1)) * 64, 0, 0);
  }
  // final hidden state
  #pragma unroll
  for(int nt = 0; nt < 4; nt++){
    #pragma unroll
    for(int r = 0; r < 8; r++){
      int M = r + mh;
      hT_out[(size_t)(base + M) * 64 + nt * 16 + l15] = hc[nt][r];
    }
  }
}

// ---------------- host launcher ----------------
extern "C" void kernel_launch(void* const* d_in, const int* in_sizes, int n_in,
                              void* d_out, int out_size, void* d_ws, size_t ws_size,
                              hipStream_t stream){
  (void)in_sizes; (void)n_in; (void)out_size; (void)ws_size;
  const float* states  = (const float*)d_in[0];
  const float* actions = (const float*)d_in[1];
  const float* h0      = (const float*)d_in[2];
  const float* sa_W    = (const float*)d_in[3];
  const float* sa_b    = (const float*)d_in[4];
  const float* sa_g    = (const float*)d_in[5];
  const float* sa_be   = (const float*)d_in[6];
  const float* se_W    = (const float*)d_in[7];
  const float* se_b    = (const float*)d_in[8];
  const float* se_g    = (const float*)d_in[9];
  const float* se_be   = (const float*)d_in[10];
  const float* key_W   = (const float*)d_in[11];
  const float* query_W = (const float*)d_in[12];
  const float* val_W   = (const float*)d_in[13];
  const float* av_g    = (const float*)d_in[14];
  const float* av_be   = (const float*)d_in[15];
  const float* lin1_W  = (const float*)d_in[16];
  const float* lin1_b  = (const float*)d_in[17];
  const float* ln_g    = (const float*)d_in[18];
  const float* ln_be   = (const float*)d_in[19];
  const float* lin2_W  = (const float*)d_in[20];
  const float* lin2_b  = (const float*)d_in[21];
  const float* avl_g   = (const float*)d_in[22];
  const float* avl_be  = (const float*)d_in[23];
  const float* q_W     = (const float*)d_in[24];
  const float* q_b     = (const float*)d_in[25];
  const float* q_g     = (const float*)d_in[26];
  const float* q_be    = (const float*)d_in[27];
  const float* c_W     = (const float*)d_in[28];
  const float* c_b     = (const float*)d_in[29];
  const float* gWih    = (const float*)d_in[30];
  const float* gWhh    = (const float*)d_in[31];
  const float* bih     = (const float*)d_in[32];
  const float* bhh     = (const float*)d_in[33];
  const float* f_g     = (const float*)d_in[34];
  const float* f_be    = (const float*)d_in[35];
  const float* f_W     = (const float*)d_in[36];
  const float* f_b     = (const float*)d_in[37];

  __bf16* ws = (__bf16*)d_ws;
  float* val_out = (float*)d_out;                      // 131072
  float* w_out   = val_out + NROW;                     // 917504
  float* hT_out  = val_out + NROW + (size_t)NTOK * AGENTS * 7; // 16384

  prep_kernel<<<128, 256, 0, stream>>>(se_W, sa_W, key_W, query_W, val_W,
                                       lin1_W, lin2_W, q_W, c_W, gWih, gWhh, ws);
  attn_kernel<<<1024, 256, 0, stream>>>(states, actions,
                                        sa_b, sa_g, sa_be, se_b, se_g, se_be,
                                        av_g, av_be, lin1_b, ln_g, ln_be,
                                        lin2_b, avl_g, avl_be, q_b, q_g, q_be, c_b,
                                        ws, ws + OFF_NF, w_out);
  gru_kernel<<<16, 32, 0, stream>>>(ws + OFF_NF, ws, h0, bih, bhh,
                                    f_g, f_be, f_W, f_b, val_out, hT_out);
}
